// HeteroSAGE_25305947308177
// MI455X (gfx1250) — matrix operand order repaired and checked by reading, not compiled, not run on hardware
//
#include <hip/hip_runtime.h>
#include <hip/hip_bf16.h>

typedef __attribute__((ext_vector_type(16))) _Float16 v16h;
typedef __attribute__((ext_vector_type(8)))  _Float16 v8h;
typedef __attribute__((ext_vector_type(8)))  float    v8f;

// ---------------------------------------------------------------------------
// Y[M,32] = X[M,K] @ W[K,32] (+ bias); K in {32,64}.
// W is pre-transposed & pre-converted to f16: WT[n*K + k]  (n = 0..31).
// One wave = one 16-row tile, two 16x16 N-tiles. M % 16 == 0 guaranteed.
// A loads: 4x b128 per K-chunk; B loads: 4x b128 (contiguous f16 columns).
// ---------------------------------------------------------------------------
__global__ void gemm_wmma_kernel(const float* __restrict__ X,
                                 const _Float16* __restrict__ WT,
                                 const float* __restrict__ bias,
                                 float* __restrict__ Y, int M, int K) {
    const int wave = threadIdx.x >> 5;
    const int lane = threadIdx.x & 31;
    const int tile = blockIdx.x * (blockDim.x >> 5) + wave;
    if (tile * 16 >= M) return;                 // wave-uniform: EXEC stays all-1s

    const int m     = lane & 15;
    const int half_ = lane >> 4;

    v8f c0 = {}, c1 = {};
    const int rowbase = tile * 16;

    for (int kc = 0; kc < K; kc += 32) {
        // ---- A tile 16x32 f16 (ISA 7.12.2): lane m holds row rowbase+m ----
        // elems 0..7  -> K = kc + 8*half + e ; elems 8..15 -> K = kc+16+8*half+e
        const float4* xv = (const float4*)(X + (size_t)(rowbase + m) * K + kc);
        const float4 x0 = xv[2 * half_];
        const float4 x1 = xv[2 * half_ + 1];
        const float4 x2 = xv[4 + 2 * half_];
        const float4 x3 = xv[5 + 2 * half_];
        v16h a;
        a[0]  = (_Float16)x0.x; a[1]  = (_Float16)x0.y;
        a[2]  = (_Float16)x0.z; a[3]  = (_Float16)x0.w;
        a[4]  = (_Float16)x1.x; a[5]  = (_Float16)x1.y;
        a[6]  = (_Float16)x1.z; a[7]  = (_Float16)x1.w;
        a[8]  = (_Float16)x2.x; a[9]  = (_Float16)x2.y;
        a[10] = (_Float16)x2.z; a[11] = (_Float16)x2.w;
        a[12] = (_Float16)x3.x; a[13] = (_Float16)x3.y;
        a[14] = (_Float16)x3.z; a[15] = (_Float16)x3.w;

        // ---- B tile 32x16 f16: lane (m,half) holds col m, K = kc+8*half+e / kc+16+8*half+e --
        // WT is [32,K] so the 16 halves are contiguous -> two b128 per column.
        const v8h* w0 = (const v8h*)(WT + (size_t)m * K + kc);
        const v8h* w1 = (const v8h*)(WT + (size_t)(16 + m) * K + kc);
        const v8h p0 = w0[half_], p1 = w0[2 + half_], q0 = w1[half_], q1 = w1[2 + half_];
        v16h b0, b1;
        #pragma unroll
        for (int e = 0; e < 8; ++e) {
            b0[e] = p0[e]; b0[8 + e] = p1[e];
            b1[e] = q0[e]; b1[8 + e] = q1[e];
        }

        c0 = __builtin_amdgcn_wmma_f32_16x16x32_f16(false, a, false, b0,
                                                    (short)0, c0, false, false);
        c1 = __builtin_amdgcn_wmma_f32_16x16x32_f16(false, a, false, b1,
                                                    (short)0, c1, false, false);
    }

    float bias0 = 0.f, bias1 = 0.f;
    if (bias) { bias0 = bias[m]; bias1 = bias[16 + m]; }

    // C/D layout: vgpr r holds M = 8*half + r, N = m (per 16-col tile)
    #pragma unroll
    for (int r = 0; r < 8; ++r) {
        const int row = rowbase + 8 * half_ + r;
        Y[(size_t)row * 32 + m]      = c0[r] + bias0;
        Y[(size_t)row * 32 + 16 + m] = c1[r] + bias1;
    }
}

// ---------------------------------------------------------------------------
// Edge scatter: agg[dst] += Y[src] (32 floats), cnt[dst] += 1.  8 lanes/edge,
// float4 per lane -> global_atomic_add_f32 x4 (exact f32 accumulation).
// ---------------------------------------------------------------------------
__global__ void scatter_add_kernel(const float* __restrict__ Y,
                                   const int* __restrict__ src,
                                   const int* __restrict__ dst,
                                   float* __restrict__ agg,
                                   float* __restrict__ cnt, int E) {
    const long long t = (long long)blockIdx.x * blockDim.x + threadIdx.x;
    const long long e = t >> 3;
    const int part = (int)(t & 7);
    if (e >= E) return;
    const int s = src[e], d = dst[e];
    const float4 v = *(const float4*)(Y + (size_t)s * 32 + part * 4);
    float* out = agg + (size_t)d * 32 + part * 4;
    atomicAdd(out + 0, v.x);
    atomicAdd(out + 1, v.y);
    atomicAdd(out + 2, v.z);
    atomicAdd(out + 3, v.w);
    if (cnt && part == 0) atomicAdd(cnt + d, 1.0f);
}

// out = base + aggA/max(cntA,1) [+ aggB/max(cntB,1)], optional relu
__global__ void combine_kernel(const float* __restrict__ aggA,
                               const float* __restrict__ cntA,
                               const float* __restrict__ aggB,
                               const float* __restrict__ cntB,
                               const float* __restrict__ base,
                               float* __restrict__ out,
                               int nrows, int do_relu) {
    const long long i = (long long)blockIdx.x * blockDim.x + threadIdx.x;
    if (i >= (long long)nrows * 32) return;
    const long long row = i >> 5;
    float r = base[i] + aggA[i] / fmaxf(cntA[row], 1.0f);
    if (aggB) r += aggB[i] / fmaxf(cntB[row], 1.0f);
    if (do_relu) r = fmaxf(r, 0.0f);
    out[i] = r;
}

__global__ void zero_kernel(float* __restrict__ p, long long n) {
    const long long i = (long long)blockIdx.x * blockDim.x + threadIdx.x;
    if (i < n) p[i] = 0.0f;
}

// WT[n*K + k] = f16( Wa[k*32 + n] (+ Wb[k*32 + n]) )  — transpose + convert.
__global__ void prep_weight_kernel(const float* __restrict__ Wa,
                                   const float* __restrict__ Wb,
                                   _Float16* __restrict__ WT, int K) {
    const int t = blockIdx.x * blockDim.x + threadIdx.x;
    if (t >= 32 * K) return;
    const int n = t / K, k = t % K;
    float v = Wa[(size_t)k * 32 + n];
    if (Wb) v += Wb[(size_t)k * 32 + n];
    WT[t] = (_Float16)v;
}

__global__ void bias_sum_kernel(const float* b1a, const float* b1b,
                                const float* b2a, const float* b2b,
                                float* bsum1, float* bsum2) {
    const int t = threadIdx.x;
    if (t < 32) {
        bsum1[t] = b1a[t] + b1b[t];
        bsum2[t] = b2a[t] + b2b[t];
    }
}

extern "C" void kernel_launch(void* const* d_in, const int* in_sizes, int n_in,
                              void* d_out, int out_size, void* d_ws, size_t ws_size,
                              hipStream_t stream) {
    const float* x_paper  = (const float*)d_in[0];
    const float* x_author = (const float*)d_in[1];
    const int* cites_src  = (const int*)d_in[2];
    const int* cites_dst  = (const int*)d_in[3];
    const int* writes_src = (const int*)d_in[4];
    const int* writes_dst = (const int*)d_in[5];
    const int* rev_src    = (const int*)d_in[6];
    const int* rev_dst    = (const int*)d_in[7];
    const float* W1l_c = (const float*)d_in[8];
    const float* W1r_c = (const float*)d_in[9];
    const float* b1_c  = (const float*)d_in[10];
    const float* W1l_w = (const float*)d_in[11];
    const float* W1r_w = (const float*)d_in[12];
    const float* b1_w  = (const float*)d_in[13];
    const float* W1l_r = (const float*)d_in[14];
    const float* W1r_r = (const float*)d_in[15];
    const float* b1_r  = (const float*)d_in[16];
    const float* W2l_c = (const float*)d_in[17];
    const float* W2r_c = (const float*)d_in[18];
    const float* b2_c  = (const float*)d_in[19];
    const float* W2l_w = (const float*)d_in[20];
    const float* W2r_w = (const float*)d_in[21];
    const float* b2_w  = (const float*)d_in[22];

    const int D = 64, H = 32;
    const int N_P = in_sizes[0] / D;      // 200000
    const int N_A = in_sizes[1] / D;      // 100000
    const int E   = in_sizes[2];          // 1000000
    const size_t PH = (size_t)N_P * H;
    const size_t AH = (size_t)N_A * H;

    // Workspace layout (floats), ~168 MB; layer-2 aliases layer-1 buffers.
    float* f      = (float*)d_ws;
    float* yl_c   = f;                    // layer1 cites proj   / layer2 y2_c
    float* yl_r   = f + PH;               // layer1 rev proj     / layer2 base2
    float* base_p = f + 2 * PH;           // self term           / becomes p1
    float* agg_c  = f + 3 * PH;           // cites agg (both layers)
    float* agg_w  = f + 4 * PH;           // writes agg (both layers)
    float* yl_w   = f + 5 * PH;           // layer1 writes proj  / layer2 y2_w
    float* base_a = yl_w + AH;            // author self term    / becomes a1
    float* agg_r  = base_a + AH;          // rev agg
    float* cnt_c  = agg_r + AH;
    float* cnt_w  = cnt_c + N_P;
    float* cnt_r  = cnt_w + N_P;
    float* bsum1  = cnt_r + N_A;
    float* bsum2  = bsum1 + 32;

    // f16 transposed weights (16B-aligned: all offsets are multiples of 4 floats)
    _Float16* hb     = (_Float16*)(bsum2 + 32);
    _Float16* T1l_c  = hb;                 // [32,64]
    _Float16* T1l_r  = hb + 2048;
    _Float16* T1sum  = hb + 4096;          // W1r_c + W1r_w
    _Float16* T1l_w  = hb + 6144;
    _Float16* T1r_r  = hb + 8192;
    _Float16* T2l_c  = hb + 10240;         // [32,32]
    _Float16* T2sum  = hb + 11264;         // W2r_c + W2r_w
    _Float16* T2l_w  = hb + 12288;

    const int B = 256;
    auto grid1 = [&](long long n) { return (unsigned)((n + B - 1) / B); };
    auto gemm_grid = [&](int M) { return (unsigned)(((M / 16) + 3) / 4); };

    // 0) weight prep: transpose + f32->f16 (+ dst-side fusion), bias sums
    prep_weight_kernel<<<8, B, 0, stream>>>(W1l_c, nullptr, T1l_c, 64);
    prep_weight_kernel<<<8, B, 0, stream>>>(W1l_r, nullptr, T1l_r, 64);
    prep_weight_kernel<<<8, B, 0, stream>>>(W1r_c, W1r_w,   T1sum, 64);
    prep_weight_kernel<<<8, B, 0, stream>>>(W1l_w, nullptr, T1l_w, 64);
    prep_weight_kernel<<<8, B, 0, stream>>>(W1r_r, nullptr, T1r_r, 64);
    prep_weight_kernel<<<4, B, 0, stream>>>(W2l_c, nullptr, T2l_c, 32);
    prep_weight_kernel<<<4, B, 0, stream>>>(W2r_c, W2r_w,   T2sum, 32);
    prep_weight_kernel<<<4, B, 0, stream>>>(W2l_w, nullptr, T2l_w, 32);
    bias_sum_kernel<<<1, 32, 0, stream>>>(b1_c, b1_w, b2_c, b2_w, bsum1, bsum2);

    // 1) zero aggregation + count buffers
    zero_kernel<<<grid1(2 * PH), B, 0, stream>>>(agg_c, (long long)(2 * PH));
    zero_kernel<<<grid1(AH + 2 * N_P + N_A), B, 0, stream>>>(
        agg_r, (long long)(AH + 2 * N_P + N_A));

    // 2) layer-1 GEMMs (project BEFORE scatter: halves edge traffic)
    gemm_wmma_kernel<<<gemm_grid(N_P), 128, 0, stream>>>(x_paper,  T1l_c, nullptr, yl_c,   N_P, D);
    gemm_wmma_kernel<<<gemm_grid(N_P), 128, 0, stream>>>(x_paper,  T1l_r, nullptr, yl_r,   N_P, D);
    gemm_wmma_kernel<<<gemm_grid(N_P), 128, 0, stream>>>(x_paper,  T1sum, bsum1,   base_p, N_P, D);
    gemm_wmma_kernel<<<gemm_grid(N_A), 128, 0, stream>>>(x_author, T1l_w, nullptr, yl_w,   N_A, D);
    gemm_wmma_kernel<<<gemm_grid(N_A), 128, 0, stream>>>(x_author, T1r_r, b1_r,    base_a, N_A, D);

    // 3) layer-1 scatters (counts computed here, reused in layer 2)
    const unsigned sgrid = grid1((long long)E * 8);
    scatter_add_kernel<<<sgrid, B, 0, stream>>>(yl_c, cites_src,  cites_dst,  agg_c, cnt_c, E);
    scatter_add_kernel<<<sgrid, B, 0, stream>>>(yl_w, writes_src, writes_dst, agg_w, cnt_w, E);
    scatter_add_kernel<<<sgrid, B, 0, stream>>>(yl_r, rev_src,    rev_dst,    agg_r, cnt_r, E);

    // 4) combine + relu -> p1 (in base_p), a1 (in base_a)
    combine_kernel<<<grid1(PH), B, 0, stream>>>(agg_c, cnt_c, agg_w, cnt_w,
                                                base_p, base_p, N_P, 1);
    combine_kernel<<<grid1(AH), B, 0, stream>>>(agg_r, cnt_r, nullptr, nullptr,
                                                base_a, base_a, N_A, 1);

    // 5) re-zero paper agg buffers for layer 2
    zero_kernel<<<grid1(2 * PH), B, 0, stream>>>(agg_c, (long long)(2 * PH));

    // 6) layer-2 GEMMs (K = H = 32)
    gemm_wmma_kernel<<<gemm_grid(N_P), 128, 0, stream>>>(base_p, T2l_c, nullptr, yl_c, N_P, H);
    gemm_wmma_kernel<<<gemm_grid(N_P), 128, 0, stream>>>(base_p, T2sum, bsum2,   yl_r, N_P, H);
    gemm_wmma_kernel<<<gemm_grid(N_A), 128, 0, stream>>>(base_a, T2l_w, nullptr, yl_w, N_A, H);

    // 7) layer-2 scatters (reuse layer-1 counts)
    scatter_add_kernel<<<sgrid, B, 0, stream>>>(yl_c, cites_src,  cites_dst,  agg_c, nullptr, E);
    scatter_add_kernel<<<sgrid, B, 0, stream>>>(yl_w, writes_src, writes_dst, agg_w, nullptr, E);

    // 8) final combine (no relu) -> d_out
    combine_kernel<<<grid1(PH), B, 0, stream>>>(agg_c, cnt_c, agg_w, cnt_w,
                                                yl_r, (float*)d_out, N_P, 0);
}
